// DocREModel_21732534517795
// MI455X (gfx1250) — matrix operand order repaired and checked by reading, not compiled, not run on hardware
//
#include <hip/hip_runtime.h>
#include <hip/hip_bf16.h>

#define B_    4
#define L_    1024
#define D_    768
#define H_    12
#define NE_   42
#define M_    8
#define P_    (NE_ * (NE_ - 1))   // 1722
#define PPAD_ 1728                // 54 * 32
#define OFF_  1
#define EPS_  1e-5f

typedef __attribute__((ext_vector_type(16))) _Float16 v16h;
typedef __attribute__((ext_vector_type(8)))  _Float16 v8h;
typedef __attribute__((ext_vector_type(8)))  float    v8f;
typedef int i32x2 __attribute__((ext_vector_type(2)));
typedef int i32x4 __attribute__((ext_vector_type(4)));

#define GAS __attribute__((address_space(1)))
#define LAS __attribute__((address_space(3)))

#if defined(__gfx1250__) &&                                              \
    __has_builtin(__builtin_amdgcn_global_load_async_to_lds_b128) &&     \
    __has_builtin(__builtin_amdgcn_global_load_async_to_lds_b64) &&      \
    __has_builtin(__builtin_amdgcn_s_wait_asynccnt)
#define USE_ASYNC_LDS 1
#else
#define USE_ASYNC_LDS 0
#endif

// ---------------------------------------------------------------------------
// K1: entity_att[b,h,e,l] = sum_m mask[b,e,m]*att[b,h,pos+1,l] / sum_m mask
// ---------------------------------------------------------------------------
__global__ __launch_bounds__(256)
void k_entity_att(const float* __restrict__ att, const float* __restrict__ mask,
                  const int* __restrict__ pos, float* __restrict__ ea) {
    int blk = blockIdx.x;               // b*H*NE + h*NE + e
    int e = blk % NE_;
    int h = (blk / NE_) % H_;
    int b = blk / (NE_ * H_);
    float mk[M_]; int id[M_];
    float msum = 0.f;
#pragma unroll
    for (int m = 0; m < M_; ++m) {
        mk[m] = mask[((size_t)b * NE_ + e) * M_ + m];
        id[m] = pos[((size_t)b * NE_ + e) * M_ + m] + OFF_;
        msum += mk[m];
    }
    float inv = 1.0f / msum;
    const float* abase = att + ((size_t)b * H_ + h) * (size_t)L_ * L_;
    float* obase = ea + (((size_t)b * H_ + h) * NE_ + e) * (size_t)L_;
    for (int l = threadIdx.x; l < L_; l += 256) {
        float acc = 0.f;
#pragma unroll
        for (int m = 0; m < M_; ++m)
            if (mk[m] != 0.f)
                acc += mk[m] * abase[(size_t)id[m] * L_ + l];
        obase[l] = acc * inv;
    }
}

// ---------------------------------------------------------------------------
// K2: entity_embed = masked logsumexp over M of gathered embeddings
// ---------------------------------------------------------------------------
__global__ __launch_bounds__(256)
void k_entity_embed(const float* __restrict__ so, const float* __restrict__ mask,
                    const int* __restrict__ pos, float* __restrict__ out) {
    int blk = blockIdx.x;               // b*NE + e
    int e = blk % NE_;
    int b = blk / NE_;
    float mk[M_]; int id[M_];
#pragma unroll
    for (int m = 0; m < M_; ++m) {
        mk[m] = mask[((size_t)b * NE_ + e) * M_ + m];
        id[m] = pos[((size_t)b * NE_ + e) * M_ + m] + OFF_;
    }
    const float* sb = so + (size_t)b * L_ * D_;
    float* ob = out + ((size_t)b * NE_ + e) * (size_t)D_;
    for (int d = threadIdx.x; d < D_; d += 256) {
        float v[M_];
        float mx = -3.4028234663852886e38f;
#pragma unroll
        for (int m = 0; m < M_; ++m) {
            v[m] = sb[(size_t)id[m] * D_ + d];
            if (mk[m] > 0.f) mx = fmaxf(mx, v[m]);
        }
        float s = 0.f;
#pragma unroll
        for (int m = 0; m < M_; ++m)
            if (mk[m] > 0.f) s += expf(v[m] - mx);
        ob[d] = mx + logf(s);
    }
}

// ---------------------------------------------------------------------------
// K3: transpose sequence_output (L x D) -> f16 (D x L) per batch
// ---------------------------------------------------------------------------
__global__ void k_transpose_f16(const float* __restrict__ so, _Float16* __restrict__ st) {
    __shared__ float tile[32][33];
    int b = blockIdx.z;
    int l0 = blockIdx.x * 32;
    int d0 = blockIdx.y * 32;
    int tx = threadIdx.x, ty = threadIdx.y;     // (32, 8)
    const float* sb = so + (size_t)b * L_ * D_;
    _Float16* ob = st + (size_t)b * D_ * L_;
    for (int i = ty; i < 32; i += 8)
        tile[i][tx] = sb[(size_t)(l0 + i) * D_ + d0 + tx];
    __syncthreads();
    for (int i = ty; i < 32; i += 8)
        ob[(size_t)(d0 + i) * L_ + l0 + tx] = (_Float16)tile[tx][i];
}

// ---------------------------------------------------------------------------
// K4: A[b,p,l] = normalize_l( (1/H) sum_h ea[h,i,l]*ea[h,j,l] ) -> f16
//     rows p >= P are zero padding
// ---------------------------------------------------------------------------
__global__ __launch_bounds__(256)
void k_pair_rows(const float* __restrict__ ea, const int* __restrict__ hts,
                 _Float16* __restrict__ A) {
    int p = blockIdx.x;                 // 0..PPAD-1
    int b = blockIdx.y;
    int tid = threadIdx.x;
    _Float16* arow = A + ((size_t)b * PPAD_ + p) * (size_t)L_;
    if (p >= P_) {
        for (int l = tid; l < L_; l += 256) arow[l] = (_Float16)0.f;
        return;
    }
    __shared__ float red[8];
    int i = hts[(((size_t)b * P_) + p) * 2 + 0];
    int j = hts[(((size_t)b * P_) + p) * 2 + 1];
    const float* eb = ea + (size_t)b * H_ * NE_ * L_;
    float vals[L_ / 256];
    float lsum = 0.f;
#pragma unroll
    for (int t = 0; t < L_ / 256; ++t) {
        int l = tid + t * 256;
        float s = 0.f;
#pragma unroll
        for (int h = 0; h < H_; ++h)
            s += eb[((size_t)h * NE_ + i) * L_ + l] * eb[((size_t)h * NE_ + j) * L_ + l];
        s *= (1.0f / H_);
        vals[t] = s;
        lsum += s;
    }
    for (int off = 16; off > 0; off >>= 1) lsum += __shfl_xor(lsum, off);
    if ((tid & 31) == 0) red[tid >> 5] = lsum;
    __syncthreads();
    float total = red[0] + red[1] + red[2] + red[3] + red[4] + red[5] + red[6] + red[7];
    float invden = 1.0f / (total + EPS_);
#pragma unroll
    for (int t = 0; t < L_ / 256; ++t)
        arow[tid + t * 256] = (_Float16)(vals[t] * invden);
}

// ---------------------------------------------------------------------------
// K5: GEMM  C[b,p,d] = sum_l A16[b,p,l] * ST16[b,d,l]  via v_wmma_f32_16x16x32_f16
//     block = 256 thr (8 waves): 32(M) x 256(N) tile; wave = 16(M) x 64(N)
//     double-buffered LDS, async global->LDS staging, split even/odd
//     accumulators so each WMMA accumulates strictly in place (no pipeline
//     copies / hazard NOPs)
// ---------------------------------------------------------------------------
#define APITCH 40   // halves; 80B rows: 16B-aligned, bank-spread
#define SPITCH 40
#define NSTEP  (L_ / 32)

__global__ __launch_bounds__(256)
void k_gemm_wmma(const _Float16* __restrict__ A, const _Float16* __restrict__ ST,
                 float* __restrict__ out) {
    __shared__ __align__(16) _Float16 Als[2][32 * APITCH];    // 2 x 2560 B
    __shared__ __align__(16) _Float16 Sls[2][256 * SPITCH];   // 2 x 20480 B
    const int b = blockIdx.z;
    const int m_blk = blockIdx.x * 32;
    const int n_blk = blockIdx.y * 256;
    const int tid  = threadIdx.x;
    const int lane = tid & 31;
    const int wave = tid >> 5;
    const int mw = wave & 1;                  // 0..1 -> M sub-tile
    const int nw = wave >> 1;                 // 0..3 -> N sub-tile (64 wide)
    const int half = lane >> 4;
    const int l16  = lane & 15;

    // staging source addresses (fixed per thread; k0 added per step)
    const int ar = tid >> 3;                  // A tile row 0..31
    const int ac = (tid & 7) * 4;             // 4 halves (8B) per thread
    const _Float16* a_src = A  + ((size_t)b * PPAD_ + m_blk + ar) * L_ + ac;
    const _Float16* s_src = ST + ((size_t)b * D_ + n_blk + tid) * L_;

    auto stage = [&](int bufi, int k0) {
#if USE_ASYNC_LDS
        __builtin_amdgcn_global_load_async_to_lds_b64(
            (GAS i32x2*)(a_src + k0), (LAS i32x2*)&Als[bufi][ar * APITCH + ac], 0, 0);
        const _Float16* s = s_src + k0;
        _Float16* d = &Sls[bufi][tid * SPITCH];
        __builtin_amdgcn_global_load_async_to_lds_b128((GAS i32x4*)(s +  0), (LAS i32x4*)(d +  0), 0, 0);
        __builtin_amdgcn_global_load_async_to_lds_b128((GAS i32x4*)(s +  8), (LAS i32x4*)(d +  8), 0, 0);
        __builtin_amdgcn_global_load_async_to_lds_b128((GAS i32x4*)(s + 16), (LAS i32x4*)(d + 16), 0, 0);
        __builtin_amdgcn_global_load_async_to_lds_b128((GAS i32x4*)(s + 24), (LAS i32x4*)(d + 24), 0, 0);
#else
        *(uint2*)(&Als[bufi][ar * APITCH + ac]) = *(const uint2*)(a_src + k0);
        const _Float16* s = s_src + k0;
        _Float16* d = &Sls[bufi][tid * SPITCH];
        *(uint4*)(d +  0) = *(const uint4*)(s +  0);
        *(uint4*)(d +  8) = *(const uint4*)(s +  8);
        *(uint4*)(d + 16) = *(const uint4*)(s + 16);
        *(uint4*)(d + 24) = *(const uint4*)(s + 24);
#endif
    };

    // fragment offsets in LDS (halves), constant per thread
    const int a_off  = (mw * 16 + l16) * APITCH + half * 16;
    const int s_off0 = (nw * 64 + l16) * SPITCH + half * 16;

    auto compute = [&](int bufi, v8f (&cc)[4]) {
        const _Float16* Ac = &Als[bufi][0];
        const _Float16* Sc = &Sls[bufi][0];
        // A fragment (contiguous K, matching B):
        // lanes 0-15:  K0..15
        // lanes 16-31: K16..31
        v8h alo = *(const v8h*)(Ac + a_off);
        v8h ahi = *(const v8h*)(Ac + a_off + 8);
        v16h a;
#pragma unroll
        for (int q = 0; q < 8; ++q) { a[q] = alo[q]; a[q + 8] = ahi[q]; }
        // B fragments: lane n holds column n, K half*16..+15 contiguous in S^T row
#pragma unroll
        for (int f = 0; f < 4; ++f) {
            const _Float16* sp = Sc + s_off0 + f * 16 * SPITCH;
            v8h blo = *(const v8h*)(sp);
            v8h bhi = *(const v8h*)(sp + 8);
            v16h bf;
#pragma unroll
            for (int q = 0; q < 8; ++q) { bf[q] = blo[q]; bf[q + 8] = bhi[q]; }
            cc[f] = __builtin_amdgcn_wmma_f32_16x16x32_f16(
                false, a, false, bf, (short)0, cc[f], false, false);
        }
    };

    v8f ce[4] = {v8f{}, v8f{}, v8f{}, v8f{}};   // even k-steps (buffer 0)
    v8f co[4] = {v8f{}, v8f{}, v8f{}, v8f{}};   // odd  k-steps (buffer 1)

    stage(0, 0);
#if USE_ASYNC_LDS
    __builtin_amdgcn_s_wait_asynccnt(0);
#endif
    __syncthreads();

    for (int step = 0; step < NSTEP; step += 2) {
        stage(1, (step + 1) * 32);              // step+1 <= 31 always valid
        compute(0, ce);
#if USE_ASYNC_LDS
        __builtin_amdgcn_s_wait_asynccnt(0);
#endif
        __syncthreads();

        if (step + 2 < NSTEP) stage(0, (step + 2) * 32);
        compute(1, co);
#if USE_ASYNC_LDS
        __builtin_amdgcn_s_wait_asynccnt(0);
#endif
        __syncthreads();
    }

    // C layout: VGPR r -> M = r (lanes 0-15) / 8+r (lanes 16-31); N = lane&15
    int row0 = m_blk + mw * 16 + half * 8;
#pragma unroll
    for (int f = 0; f < 4; ++f) {
        v8f cf = ce[f] + co[f];
        int col = n_blk + nw * 64 + f * 16 + l16;
#pragma unroll
        for (int r = 0; r < 8; ++r) {
            int row = row0 + r;
            if (row < P_)
                out[((size_t)b * P_ + row) * D_ + col] = cf[r];
        }
    }
}

// ---------------------------------------------------------------------------
extern "C" void kernel_launch(void* const* d_in, const int* in_sizes, int n_in,
                              void* d_out, int out_size, void* d_ws, size_t ws_size,
                              hipStream_t stream) {
    const float* so   = (const float*)d_in[0];   // (B,L,D)
    const float* att  = (const float*)d_in[1];   // (B,H,L,L)
    const float* mask = (const float*)d_in[2];   // (B,NE,M)
    const int*   pos  = (const int*)d_in[3];     // (B,NE,M)
    const int*   hts  = (const int*)d_in[4];     // (B,P,2)

    float* out_emb = (float*)d_out;                          // (B,NE,D)
    float* out_ht  = out_emb + (size_t)B_ * NE_ * D_;        // (B*P,D)

    constexpr size_t EA_BYTES = (size_t)B_ * H_ * NE_ * L_ * sizeof(float);     // 8,257,536
    constexpr size_t ST_BYTES = (size_t)B_ * D_ * L_ * sizeof(_Float16);        // 6,291,456
    char* ws = (char*)d_ws;
    float*    ws_ea = (float*)ws;
    _Float16* ws_st = (_Float16*)(ws + EA_BYTES);
    _Float16* ws_a  = (_Float16*)(ws + EA_BYTES + ST_BYTES);

    k_entity_att  <<<B_ * H_ * NE_, 256, 0, stream>>>(att, mask, pos, ws_ea);
    k_entity_embed<<<B_ * NE_,      256, 0, stream>>>(so, mask, pos, out_emb);
    k_transpose_f16<<<dim3(L_ / 32, D_ / 32, B_), dim3(32, 8), 0, stream>>>(so, ws_st);
    k_pair_rows   <<<dim3(PPAD_, B_), 256, 0, stream>>>(ws_ea, hts, ws_a);
    k_gemm_wmma   <<<dim3(PPAD_ / 32, D_ / 256, B_), 256, 0, stream>>>(ws_a, ws_st, out_ht);
}
